// GlobalSparseMoeBlock_49821620633678
// MI455X (gfx1250) — compile-verified, matrix operand order checked by reading, not run on hardware
//
#include <hip/hip_runtime.h>
#include <hip/hip_bf16.h>
#include <math.h>

// Problem constants (from reference)
#define NTOK 2048   // B*T
#define HD   768    // hidden
#define NE   64     // experts
#define ID   256    // intermediate
#define TK   8      // top-k
#define NK   (NTOK*TK)

#define MT   64     // M rows per workgroup
#define KT   32     // K per WMMA step (bf16)

typedef __attribute__((ext_vector_type(16))) __bf16 v16bf;
typedef __attribute__((ext_vector_type(8)))  __bf16 v8bf;
typedef __attribute__((ext_vector_type(8)))  float  v8f;
typedef __attribute__((ext_vector_type(4)))  unsigned int u32x4;
typedef __attribute__((ext_vector_type(8)))  int i32x8;
typedef __attribute__((ext_vector_type(4)))  int i32x4;

__device__ __forceinline__ unsigned short f2bfu(float f) {
    unsigned u = __builtin_bit_cast(unsigned, f);
    u += 0x7FFFu + ((u >> 16) & 1u);   // round-to-nearest-even
    return (unsigned short)(u >> 16);
}

// ---- Tensor Data Mover: load a 2D bf16 tile [tile_d1 rows x tile_d0 elems]
// (row pitch = stride_elems) from global into contiguous LDS at lds_off.
// D# packing per CDNA5 ISA ch.8 (group0: count/lds/global/type, group1:
// data_size=2B, dims, strides). Groups 2/3 zero (<=2D tensor).
__device__ __forceinline__ void tdm_load_2d(unsigned lds_off, const void* gptr,
                                            unsigned tile_d0, unsigned tile_d1,
                                            unsigned stride_elems)
{
    unsigned long long ga = (unsigned long long)(uintptr_t)gptr;
    u32x4 g0;
    g0[0] = 1u;                                            // count=1, user desc
    g0[1] = lds_off;                                       // lds_addr (bytes)
    g0[2] = (unsigned)(ga & 0xFFFFFFFFu);                  // global_addr[31:0]
    g0[3] = (unsigned)((ga >> 32) & 0x01FFFFFFu) | (2u << 30); // addr[56:32] | type=2

    i32x8 g1;
    g1[0] = (int)(1u << 16);                               // data_size=1 (2 bytes)
    g1[1] = (int)((tile_d0 & 0xFFFFu) << 16);              // tensor_dim0[15:0] @bits63:48
    g1[2] = (int)(((tile_d0 >> 16) & 0xFFFFu)              // tensor_dim0[31:16]
                  | ((tile_d1 & 0xFFFFu) << 16));          // tensor_dim1[15:0]
    g1[3] = (int)(((tile_d1 >> 16) & 0xFFFFu)              // tensor_dim1[31:16]
                  | ((tile_d0 & 0xFFFFu) << 16));          // tile_dim0
    g1[4] = (int)(tile_d1 & 0xFFFFu);                      // tile_dim1 (tile_dim2=0)
    g1[5] = (int)stride_elems;                             // tensor_dim0_stride[31:0]
    g1[6] = 0;                                             // stride hi / dim1_stride lo
    g1[7] = 0;

    i32x4 z4 = (i32x4)0;
#if defined(__clang_major__) && (__clang_major__ >= 23)
    i32x8 z8 = (i32x8)0;
    __builtin_amdgcn_tensor_load_to_lds(g0, g1, z4, z4, z8, 0);
#else
    __builtin_amdgcn_tensor_load_to_lds(g0, g1, z4, z4, 0);
#endif
}

// LDS byte offset of a __shared__ pointer (flat shared addr low 32 bits == LDS offset)
__device__ __forceinline__ unsigned lds_offset(const void* p) {
    return (unsigned)(uintptr_t)p;
}

// Load a 16x32 bf16 fragment (A or B^T) from LDS, rows contiguous with `stride`
// shorts per row. Matches CDNA5 16-bit A/B layout:
//   lanes 0-15  : row = lane,    K chunks {0..7, 16..23}
//   lanes 16-31 : row = lane-16, K chunks {8..15, 24..31}
__device__ __forceinline__ v16bf load_frag(const unsigned short* p, int stride, int lane) {
    int r  = lane & 15;
    int kb = (lane >> 4) << 3;           // 0 or 8
    v8bf c0 = *reinterpret_cast<const v8bf*>(p + r * stride + kb);
    v8bf c1 = *reinterpret_cast<const v8bf*>(p + r * stride + kb + 16);
    v16bf out;
#pragma unroll
    for (int i = 0; i < 8; ++i) { out[i] = c0[i]; out[i + 8] = c1[i]; }
    return out;
}

__device__ __forceinline__ v8f wmma_bf16(v16bf a, v16bf b, v8f c) {
    return __builtin_amdgcn_wmma_f32_16x16x32_bf16(
        false, a, false, b, (short)0, c, false, false);
}

// ---------------- kernel 0: f32 [E][R][C] -> bf16 transposed [E][C][R] ----------------
__global__ __launch_bounds__(256) void moe_cvt_t_kernel(
    const float* __restrict__ src, unsigned short* __restrict__ dst, int R, int C)
{
    __shared__ float tile[32][33];
    int e  = blockIdx.z;
    int c0 = blockIdx.x * 32;
    int r0 = blockIdx.y * 32;
    int tx = threadIdx.x & 31;
    int ty = threadIdx.x >> 5;           // 0..7

    const float* s = src + ((size_t)e * R + r0) * C + c0;
#pragma unroll
    for (int j = 0; j < 4; ++j)
        tile[ty + j * 8][tx] = s[(size_t)(ty + j * 8) * C + tx];
    __syncthreads();
    unsigned short* d = dst + ((size_t)e * C + c0) * R + r0;
#pragma unroll
    for (int j = 0; j < 4; ++j)
        d[(size_t)(ty + j * 8) * R + tx] = f2bfu(tile[tx][ty + j * 8]);
}

// ---------------- kernel 1: zero expert counts ----------------
__global__ void moe_init_kernel(int* __restrict__ counts) {
    int i = threadIdx.x;
    if (i < NE) counts[i] = 0;
}

// ---------------- kernel 2: router (softmax -> top-8 -> renorm) ----------------
__global__ __launch_bounds__(64) void moe_router_kernel(
    const float* __restrict__ hs, const float* __restrict__ rw,
    float* __restrict__ topw, int* __restrict__ topi, int* __restrict__ counts)
{
    int t   = blockIdx.x;
    int tid = threadIdx.x;
    __shared__ float xs[HD];
    __shared__ float pr[NE];

    for (int i = tid; i < HD; i += 64) xs[i] = hs[(size_t)t * HD + i];
    __syncthreads();

    float acc = 0.f;
    const float* wrow = rw + (size_t)tid * HD;
    for (int h = 0; h < HD; ++h) acc = fmaf(xs[h], wrow[h], acc);
    pr[tid] = acc;
    __syncthreads();

    if (tid == 0) {
        float mx = pr[0];
        for (int e = 1; e < NE; ++e) mx = fmaxf(mx, pr[e]);
        float s = 0.f;
        for (int e = 0; e < NE; ++e) { pr[e] = expf(pr[e] - mx); s += pr[e]; }
        float inv = 1.f / s;
        for (int e = 0; e < NE; ++e) pr[e] *= inv;

        float tw[TK]; int ti[TK]; float wsum = 0.f;
        for (int k = 0; k < TK; ++k) {
            int am = 0; float bv = pr[0];
            for (int e = 1; e < NE; ++e) if (pr[e] > bv) { bv = pr[e]; am = e; }
            tw[k] = bv; ti[k] = am; pr[am] = -1.f; wsum += bv;
        }
        float invw = 1.f / wsum;
        for (int k = 0; k < TK; ++k) {
            topw[t * TK + k] = tw[k] * invw;
            topi[t * TK + k] = ti[k];
            atomicAdd(&counts[ti[k]], 1);
        }
    }
}

// ---------------- kernel 3: serial exclusive scan + cursor reset ----------------
__global__ void moe_scan_kernel(const int* __restrict__ counts,
                                int* __restrict__ offs, int* __restrict__ cursor)
{
    if (threadIdx.x == 0 && blockIdx.x == 0) {
        int run = 0;
        for (int e = 0; e < NE; ++e) { offs[e] = run; run += counts[e]; cursor[e] = 0; }
    }
}

// ---------------- kernel 4: scatter (token, slot, weight) into expert buckets ----------------
__global__ void moe_scatter_kernel(const float* __restrict__ topw, const int* __restrict__ topi,
                                   const int* __restrict__ offs, int* __restrict__ cursor,
                                   int* __restrict__ tok, int* __restrict__ slot,
                                   float* __restrict__ wgt)
{
    int i = blockIdx.x * blockDim.x + threadIdx.x;
    if (i < NK) {
        int t = i >> 3, k = i & 7;
        int e = topi[i];
        int pos = offs[e] + atomicAdd(&cursor[e], 1);
        tok[pos]  = t;
        slot[pos] = k;
        wgt[pos]  = topw[i];
    }
}

// ---------------- kernel 5: fused SwiGLU expert GEMM (TDM streaming + bf16 WMMA) ----------------
__global__ __launch_bounds__(256) void moe_expert_kernel(
    const float* __restrict__ hs,
    const unsigned short* __restrict__ wgT,    // [E][ID][HD] bf16
    const unsigned short* __restrict__ wuT,    // [E][ID][HD] bf16
    const unsigned short* __restrict__ wdT,    // [E][HD][ID] bf16
    const int* __restrict__ counts, const int* __restrict__ offs,
    const int* __restrict__ tok, const int* __restrict__ slot,
    const float* __restrict__ wgt, float* __restrict__ staging)
{
    const int e  = blockIdx.y;
    const int mt = blockIdx.x;
    const int cnt = counts[e];
    if (mt * MT >= cnt) return;
    const int base = offs[e] + mt * MT;
    int rows = cnt - mt * MT; if (rows > MT) rows = MT;

    __shared__ __align__(16) unsigned short smA[MT * KT];        // 4 KB A tile (bf16)
    __shared__ __align__(16) unsigned short smW[2 * ID * KT];    // 32 KB Wg^T/Wu^T, reused as hmid[MT][ID]
    __shared__ __align__(16) unsigned short smWd[ID * KT];       // 16 KB Wd^T tile
    __shared__ int   tokS[MT];
    __shared__ int   slotS[MT];
    __shared__ float wS[MT];

    const int tid  = threadIdx.x;
    const int wave = tid >> 5;
    const int lane = tid & 31;

    if (tid < MT) {
        if (tid < rows) {
            tokS[tid]  = tok[base + tid];
            slotS[tid] = slot[base + tid];
            wS[tid]    = wgt[base + tid];
        } else {
            tokS[tid] = 0; slotS[tid] = 0; wS[tid] = 0.f;
        }
    }
    __syncthreads();

    const unsigned short* wgE = wgT + (size_t)e * ID * HD;
    const unsigned short* wuE = wuT + (size_t)e * ID * HD;
    const unsigned ldsWg = lds_offset(&smW[0]);
    const unsigned ldsWu = lds_offset(&smW[ID * KT]);
    const unsigned ldsWd = lds_offset(&smWd[0]);

    v8f accG[4][2], accU[4][2];
#pragma unroll
    for (int r = 0; r < 4; ++r)
#pragma unroll
        for (int c = 0; c < 2; ++c) { accG[r][c] = (v8f)0.f; accU[r][c] = (v8f)0.f; }

    // ---- phase A: gate = X@Wg, up = X@Wu over K = HD ----
    for (int k0 = 0; k0 < HD; k0 += KT) {
        __syncthreads();                     // previous tiles fully consumed
        if (wave == 0) {                     // one TDM descriptor per weight tile
            tdm_load_2d(ldsWg, wgE + k0, KT, ID, HD);   // [ID rows x KT], pitch HD
            tdm_load_2d(ldsWu, wuE + k0, KT, ID, HD);
        }
        // overlap: gathered X tile [MT x KT], f32 -> bf16 (token-indexed rows)
#pragma unroll
        for (int it = 0; it < (MT * KT) / 256; ++it) {
            int i = it * 256 + tid;
            int m = i >> 5, kk = i & 31;
            smA[m * KT + kk] = f2bfu(hs[(size_t)tokS[m] * HD + k0 + kk]);
        }
        if (wave == 0) __builtin_amdgcn_s_wait_tensorcnt((short)0);
        __syncthreads();

        v16bf a[4], bg[2], bu[2];
#pragma unroll
        for (int r = 0; r < 4; ++r) a[r] = load_frag(&smA[r * 16 * KT], KT, lane);
        const int n0 = wave * 32;
#pragma unroll
        for (int c = 0; c < 2; ++c) {
            bg[c] = load_frag(&smW[(n0 + c * 16) * KT], KT, lane);
            bu[c] = load_frag(&smW[ID * KT + (n0 + c * 16) * KT], KT, lane);
        }
#pragma unroll
        for (int r = 0; r < 4; ++r)
#pragma unroll
            for (int c = 0; c < 2; ++c) {
                accG[r][c] = wmma_bf16(a[r], bg[c], accG[r][c]);
                accU[r][c] = wmma_bf16(a[r], bu[c], accU[r][c]);
            }
    }
    __syncthreads();

    // ---- silu(gate)*up -> bf16 hmid in LDS (reuse smW region) ----
    unsigned short* Hbuf = smW;   // [MT][ID] = 16384 shorts, exactly fits
#pragma unroll
    for (int r = 0; r < 4; ++r)
#pragma unroll
        for (int c = 0; c < 2; ++c)
#pragma unroll
            for (int vg = 0; vg < 8; ++vg) {
                int m = r * 16 + vg + ((lane >> 4) << 3);
                int n = wave * 32 + c * 16 + (lane & 15);
                float g = accG[r][c][vg];
                float u = accU[r][c][vg];
                float h = (g / (1.f + expf(-g))) * u;
                Hbuf[m * ID + n] = f2bfu(h);
            }
    __syncthreads();

    // ---- phase B: out = hmid @ Wd over K = ID, N chunks of 256 ----
    const unsigned short* wdE = wdT + (size_t)e * HD * ID;
    for (int nc = 0; nc < HD; nc += 256) {
        v8f acc[4][2];
#pragma unroll
        for (int r = 0; r < 4; ++r)
#pragma unroll
            for (int c = 0; c < 2; ++c) acc[r][c] = (v8f)0.f;

        for (int k0 = 0; k0 < ID; k0 += KT) {
            __syncthreads();
            if (wave == 0) {
                tdm_load_2d(ldsWd, wdE + (size_t)nc * ID + k0, KT, 256, ID);
                __builtin_amdgcn_s_wait_tensorcnt((short)0);
            }
            __syncthreads();

            v16bf a2[4], b2[2];
#pragma unroll
            for (int r = 0; r < 4; ++r)
                a2[r] = load_frag(&Hbuf[(r * 16) * ID + k0], ID, lane);
#pragma unroll
            for (int c = 0; c < 2; ++c)
                b2[c] = load_frag(&smWd[(wave * 32 + c * 16) * KT], KT, lane);
#pragma unroll
            for (int r = 0; r < 4; ++r)
#pragma unroll
                for (int c = 0; c < 2; ++c)
                    acc[r][c] = wmma_bf16(a2[r], b2[c], acc[r][c]);
        }

        // staged write: each (token, slot) row written by exactly one block -> no atomics
#pragma unroll
        for (int r = 0; r < 4; ++r)
#pragma unroll
            for (int c = 0; c < 2; ++c)
#pragma unroll
                for (int vg = 0; vg < 8; ++vg) {
                    int m = r * 16 + vg + ((lane >> 4) << 3);
                    if (m < rows) {
                        int n = nc + wave * 32 + c * 16 + (lane & 15);
                        int t = tokS[m], s = slotS[m];
                        staging[((size_t)s * NTOK + t) * HD + n] = wS[m] * acc[r][c][vg];
                    }
                }
    }
}

// ---------------- kernel 6: reduce 8 staging slots into output ----------------
__global__ void moe_reduce_kernel(const float* __restrict__ staging, float* __restrict__ out)
{
    int i = blockIdx.x * blockDim.x + threadIdx.x;
    if (i < NTOK * HD) {
        float s = 0.f;
#pragma unroll
        for (int k = 0; k < TK; ++k) s += staging[(size_t)k * NTOK * HD + i];
        out[i] = s;
    }
}

extern "C" void kernel_launch(void* const* d_in, const int* in_sizes, int n_in,
                              void* d_out, int out_size, void* d_ws, size_t ws_size,
                              hipStream_t stream)
{
    const float* hs = (const float*)d_in[0];   // [N, H]
    const float* rw = (const float*)d_in[1];   // [E, H]
    const float* wg = (const float*)d_in[2];   // [E, H, I]
    const float* wu = (const float*)d_in[3];   // [E, H, I]
    const float* wd = (const float*)d_in[4];   // [E, I, H]
    float* out = (float*)d_out;

    char* w = (char*)d_ws;
    size_t off = 0;
    float* topw   = (float*)(w + off); off += (size_t)NK * 4;
    int*   topi   = (int*)  (w + off); off += (size_t)NK * 4;
    int*   counts = (int*)  (w + off); off += (size_t)NE * 4;
    int*   offs   = (int*)  (w + off); off += (size_t)NE * 4;
    int*   cursor = (int*)  (w + off); off += (size_t)NE * 4;
    int*   tok    = (int*)  (w + off); off += (size_t)NK * 4;
    int*   slot   = (int*)  (w + off); off += (size_t)NK * 4;
    float* wgt    = (float*)(w + off); off += (size_t)NK * 4;
    off = (off + 255) & ~(size_t)255;
    unsigned short* wgTb = (unsigned short*)(w + off); off += (size_t)NE * ID * HD * 2;
    unsigned short* wuTb = (unsigned short*)(w + off); off += (size_t)NE * ID * HD * 2;
    unsigned short* wdTb = (unsigned short*)(w + off); off += (size_t)NE * HD * ID * 2;
    off = (off + 255) & ~(size_t)255;
    float* staging = (float*)(w + off);        // [TK, NTOK, HD] f32 (~50.3 MB)

    // one-time weight convert+transpose: f32 [R,C] -> bf16 [C,R] per expert
    moe_cvt_t_kernel<<<dim3(ID / 32, HD / 32, NE), 256, 0, stream>>>(wg, wgTb, HD, ID);
    moe_cvt_t_kernel<<<dim3(ID / 32, HD / 32, NE), 256, 0, stream>>>(wu, wuTb, HD, ID);
    moe_cvt_t_kernel<<<dim3(HD / 32, ID / 32, NE), 256, 0, stream>>>(wd, wdTb, ID, HD);

    moe_init_kernel<<<1, 64, 0, stream>>>(counts);
    moe_router_kernel<<<NTOK, 64, 0, stream>>>(hs, rw, topw, topi, counts);
    moe_scan_kernel<<<1, 1, 0, stream>>>(counts, offs, cursor);
    moe_scatter_kernel<<<(NK + 255) / 256, 256, 0, stream>>>(topw, topi, offs, cursor,
                                                             tok, slot, wgt);
    dim3 grid(NTOK / MT, NE);   // worst case: all tokens on one expert
    moe_expert_kernel<<<grid, 256, 0, stream>>>(hs, wgTb, wuTb, wdTb, counts, offs,
                                                tok, slot, wgt, staging);
    moe_reduce_kernel<<<(NTOK * HD + 255) / 256, 256, 0, stream>>>(staging, out);
}
